// DecoderLayer_44487271252363
// MI455X (gfx1250) — matrix operand order repaired and checked by reading, not compiled, not run on hardware
//
#include <hip/hip_runtime.h>
#include <hip/hip_bf16.h>

// ---------------- problem constants (match reference) ----------------
#define BB    2
#define TT    2048
#define EMB   2048
#define NH    16
#define NKV   4
#define NREP  4
#define HD    128
#define VHD   128
#define WIN   1024
#define MLPD  8192
#define MBT   (BB*TT)          // 4096 rows
#define SCALING 0.08838834764831845f   // 128^-0.5

// ---------------- types ----------------
typedef __attribute__((ext_vector_type(16))) __bf16          v16bf;
typedef __attribute__((ext_vector_type(8)))  float           v8f;
typedef __attribute__((ext_vector_type(8)))  unsigned short  v8u;
typedef __attribute__((ext_vector_type(16))) unsigned short  v16u;

static __device__ __forceinline__ unsigned short f2bf(float f) {
    unsigned u = __builtin_bit_cast(unsigned, f);
    u += 0x7fffu + ((u >> 16) & 1u);          // round-to-nearest-even
    return (unsigned short)(u >> 16);
}

static __device__ __forceinline__ v16bf mk16u(v8u lo, v8u hi) {
    v16u r;
#pragma unroll
    for (int i = 0; i < 8; ++i) { r[i] = lo[i]; r[8 + i] = hi[i]; }
    return __builtin_bit_cast(v16bf, r);
}

static __device__ __forceinline__ v8f wmma_bf16(v16bf a, v16bf b, v8f c) {
    return __builtin_amdgcn_wmma_f32_16x16x32_bf16(false, a, false, b, (short)0, c, false, false);
}

// ---------------- elementwise: f32 -> bf16 cast ----------------
__global__ __launch_bounds__(256) void cast_bf16_kernel(const float* __restrict__ in,
                                                        unsigned short* __restrict__ out,
                                                        size_t n) {
    size_t i = (size_t)blockIdx.x * blockDim.x + threadIdx.x;
    if (i < n) out[i] = f2bf(in[i]);
}

// ---------------- RMSNorm (f32 in, bf16 out), one block per row ----------------
__global__ __launch_bounds__(256) void rmsnorm_bf16_kernel(const float* __restrict__ x,
                                                           const float* __restrict__ scale,
                                                           unsigned short* __restrict__ out) {
    __shared__ float red[8];
    const int row = blockIdx.x, tid = threadIdx.x;
    const float* xr = x + (size_t)row * EMB;
    float ss = 0.f;
    for (int i = tid; i < EMB; i += 256) { float v = xr[i]; ss += v * v; }
#pragma unroll
    for (int off = 1; off < 32; off <<= 1) ss += __shfl_xor(ss, off, 32);
    if ((tid & 31) == 0) red[tid >> 5] = ss;
    __syncthreads();
    float tot = 0.f;
#pragma unroll
    for (int w = 0; w < 8; ++w) tot += red[w];
    const float r = rsqrtf(tot / (float)EMB + 1e-6f);
    for (int i = tid; i < EMB; i += 256)
        out[(size_t)row * EMB + i] = f2bf(xr[i] * r * (1.0f + scale[i]));
}

// ---------------- partial RoPE (first 64 dims) + scale + bf16 cast ----------------
__global__ __launch_bounds__(256) void rope_cast_kernel(const float* __restrict__ in,
                                                        unsigned short* __restrict__ out,
                                                        int heads, float outscale, size_t n) {
    size_t i = (size_t)blockIdx.x * blockDim.x + threadIdx.x;
    if (i >= n) return;
    const int d = (int)(i % HD);
    const size_t bt = (i / HD) / heads;
    const int t = (int)(bt % TT);
    float x = in[i], o;
    if (d < 64) {
        const int j = d & 31;
        const float inv_freq = __powf(10000.0f, -(float)j / 32.0f);
        const float ang = (float)t * inv_freq;
        const float s = __sinf(ang), c = __cosf(ang);
        const float other = (d < 32) ? in[i + 32] : in[i - 32];
        o = (d < 32) ? (x * c - other * s) : (x * c + other * s);
    } else {
        o = x;
    }
    out[i] = f2bf(o * outscale);
}

// ---------------- V: f32 [B*T][NKV*VHD] -> bf16 [B][NKV][VHD][T] (key-contiguous) ---
__global__ __launch_bounds__(256) void v_cast_transpose_kernel(const float* __restrict__ vf,
                                                               unsigned short* __restrict__ vt,
                                                               size_t n) {
    size_t i = (size_t)blockIdx.x * blockDim.x + threadIdx.x;
    if (i >= n) return;
    const int d  = (int)(i % VHD);
    size_t r = i / VHD;
    const int kv = (int)(r % NKV);
    size_t bt = r / NKV;
    const int t = (int)(bt % TT);
    const int b = (int)(bt / TT);
    vt[(((size_t)(b * NKV + kv) * VHD) + d) * TT + t] = f2bf(vf[i]);
}

// =======================================================================
// WMMA bf16 GEMM, dims compile-time: C[M,N] f32 (+resid) = A[M,K] x B[K,N]
// Block tile 128x128 (256 threads = 8 waves, 4x2), wave tile 32x64
// (2x4 v_wmma_f32_16x16x32_bf16 accumulators, 8 WMMAs per K-step).
// =======================================================================
template <int M, int N, int K, bool HAS_RESID>
__global__ __launch_bounds__(256) void gemm_kernel(const unsigned short* __restrict__ A,
                                                   const unsigned short* __restrict__ Bm,
                                                   const float* __restrict__ resid,
                                                   float* __restrict__ C) {
    __shared__ unsigned short ldsA[128 * 32];
    __shared__ unsigned short ldsBt[128 * 32];
    const int tid  = threadIdx.x;
    const int lane = tid & 31, wave = tid >> 5;
    const int half = lane >> 4, l16 = lane & 15;
    constexpr int NT = N / 128;
    const int bm = (blockIdx.x / NT) * 128;
    const int bn = (blockIdx.x % NT) * 128;
    const int wm = (wave >> 1) * 32;   // 0,32,64,96
    const int wn = (wave & 1) * 64;    // 0,64

    v8f acc[2][4] = {};
    for (int k0 = 0; k0 < K; k0 += 32) {
        if (k0 + 32 < K) {   // gfx1250 global_prefetch of next K tile
            __builtin_prefetch(A + (size_t)(bm + (tid >> 1)) * K + k0 + 32, 0, 0);
            __builtin_prefetch(Bm + (size_t)(k0 + 32 + (tid >> 3)) * N + bn + (tid & 7) * 16, 0, 0);
        }
        // stage A tile 128x32 (16 elems / thread)
        {
            const int r = tid >> 1, cg = (tid & 1) * 16;
            const unsigned short* src = A + (size_t)(bm + r) * K + k0 + cg;
            *(v8u*)(ldsA + r * 32 + cg)     = *(const v8u*)(src);
            *(v8u*)(ldsA + r * 32 + cg + 8) = *(const v8u*)(src + 8);
        }
        // stage B tile 32x128, transposed so fragments are K-contiguous
        {
            const int kr = tid >> 3, ng = (tid & 7) * 16;
            const unsigned short* src = Bm + (size_t)(k0 + kr) * N + bn + ng;
            v8u b0 = *(const v8u*)(src);
            v8u b1 = *(const v8u*)(src + 8);
#pragma unroll
            for (int i = 0; i < 8; ++i) ldsBt[(ng + i) * 32 + kr] = b0[i];
#pragma unroll
            for (int i = 0; i < 8; ++i) ldsBt[(ng + 8 + i) * 32 + kr] = b1[i];
        }
        __syncthreads();

        v16bf afr[2];
#pragma unroll
        for (int i = 0; i < 2; ++i) {
            const v8u* rowp = (const v8u*)(ldsA + (wm + i * 16 + l16) * 32);
            afr[i] = mk16u(rowp[2 * half], rowp[2 * half + 1]);
        }
#pragma unroll
        for (int j = 0; j < 4; ++j) {
            const v8u* colp = (const v8u*)(ldsBt + (wn + j * 16 + l16) * 32);
            v16bf bfr = mk16u(colp[2 * half], colp[2 * half + 1]);
#pragma unroll
            for (int i = 0; i < 2; ++i) acc[i][j] = wmma_bf16(afr[i], bfr, acc[i][j]);
        }
        __syncthreads();
    }
    // epilogue: C layout M = r + 8*half, N = l16; batch resid loads per row
#pragma unroll
    for (int i = 0; i < 2; ++i) {
#pragma unroll
        for (int r = 0; r < 8; ++r) {
            const size_t row = (size_t)(bm + wm + i * 16 + r + 8 * half);
            float* crow = C + row * N + bn + wn + l16;
            if constexpr (HAS_RESID) {
                const float* rrow = resid + row * N + bn + wn + l16;
                float rv[4];
#pragma unroll
                for (int j = 0; j < 4; ++j) rv[j] = rrow[j * 16];
#pragma unroll
                for (int j = 0; j < 4; ++j) crow[j * 16] = acc[i][j][r] + rv[j];
            } else {
#pragma unroll
                for (int j = 0; j < 4; ++j) crow[j * 16] = acc[i][j][r];
            }
        }
    }
}

// =======================================================================
// Fused gate/up GEMM + SiLU: T[M,MLPD] bf16 = silu(A x G) * (A x U)
// Shared A tile, two B tiles, two accumulator sets; kills ~540MB of f32
// intermediate HBM traffic vs separate gate/up/silu kernels.
// =======================================================================
__global__ __launch_bounds__(256) void gateup_silu_kernel(const unsigned short* __restrict__ A,  // [MBT][EMB]
                                                          const unsigned short* __restrict__ G,  // [EMB][MLPD]
                                                          const unsigned short* __restrict__ U,  // [EMB][MLPD]
                                                          unsigned short* __restrict__ T) {      // [MBT][MLPD]
    __shared__ unsigned short ldsA[128 * 32];
    __shared__ unsigned short ldsGt[128 * 32];
    __shared__ unsigned short ldsUt[128 * 32];
    const int tid  = threadIdx.x;
    const int lane = tid & 31, wave = tid >> 5;
    const int half = lane >> 4, l16 = lane & 15;
    constexpr int NT = MLPD / 128;
    const int bm = (blockIdx.x / NT) * 128;
    const int bn = (blockIdx.x % NT) * 128;
    const int wm = (wave >> 1) * 32;
    const int wn = (wave & 1) * 64;

    v8f accG[2][4] = {};
    v8f accU[2][4] = {};
    for (int k0 = 0; k0 < EMB; k0 += 32) {
        if (k0 + 32 < EMB) {
            __builtin_prefetch(A + (size_t)(bm + (tid >> 1)) * EMB + k0 + 32, 0, 0);
            __builtin_prefetch(G + (size_t)(k0 + 32 + (tid >> 3)) * MLPD + bn + (tid & 7) * 16, 0, 0);
            __builtin_prefetch(U + (size_t)(k0 + 32 + (tid >> 3)) * MLPD + bn + (tid & 7) * 16, 0, 0);
        }
        {
            const int r = tid >> 1, cg = (tid & 1) * 16;
            const unsigned short* src = A + (size_t)(bm + r) * EMB + k0 + cg;
            *(v8u*)(ldsA + r * 32 + cg)     = *(const v8u*)(src);
            *(v8u*)(ldsA + r * 32 + cg + 8) = *(const v8u*)(src + 8);
        }
        {
            const int kr = tid >> 3, ng = (tid & 7) * 16;
            const size_t srcoff = (size_t)(k0 + kr) * MLPD + bn + ng;
            v8u g0 = *(const v8u*)(G + srcoff);
            v8u g1 = *(const v8u*)(G + srcoff + 8);
            v8u u0 = *(const v8u*)(U + srcoff);
            v8u u1 = *(const v8u*)(U + srcoff + 8);
#pragma unroll
            for (int i = 0; i < 8; ++i) ldsGt[(ng + i) * 32 + kr] = g0[i];
#pragma unroll
            for (int i = 0; i < 8; ++i) ldsGt[(ng + 8 + i) * 32 + kr] = g1[i];
#pragma unroll
            for (int i = 0; i < 8; ++i) ldsUt[(ng + i) * 32 + kr] = u0[i];
#pragma unroll
            for (int i = 0; i < 8; ++i) ldsUt[(ng + 8 + i) * 32 + kr] = u1[i];
        }
        __syncthreads();

        v16bf afr[2];
#pragma unroll
        for (int i = 0; i < 2; ++i) {
            const v8u* rowp = (const v8u*)(ldsA + (wm + i * 16 + l16) * 32);
            afr[i] = mk16u(rowp[2 * half], rowp[2 * half + 1]);
        }
#pragma unroll
        for (int j = 0; j < 4; ++j) {
            const v8u* gp = (const v8u*)(ldsGt + (wn + j * 16 + l16) * 32);
            const v8u* up = (const v8u*)(ldsUt + (wn + j * 16 + l16) * 32);
            v16bf gfr = mk16u(gp[2 * half], gp[2 * half + 1]);
            v16bf ufr = mk16u(up[2 * half], up[2 * half + 1]);
#pragma unroll
            for (int i = 0; i < 2; ++i) {
                accG[i][j] = wmma_bf16(afr[i], gfr, accG[i][j]);
                accU[i][j] = wmma_bf16(afr[i], ufr, accU[i][j]);
            }
        }
        __syncthreads();
    }
    // epilogue: t = silu(g) * u -> bf16
#pragma unroll
    for (int i = 0; i < 2; ++i) {
#pragma unroll
        for (int r = 0; r < 8; ++r) {
            const size_t row = (size_t)(bm + wm + i * 16 + r + 8 * half);
            unsigned short* trow = T + row * MLPD + bn + wn + l16;
#pragma unroll
            for (int j = 0; j < 4; ++j) {
                const float g = accG[i][j][r];
                const float s = g / (1.0f + __expf(-g));
                trow[j * 16] = f2bf(s * accU[i][j][r]);
            }
        }
    }
}

// ---------------- flash attention with sliding window + sink ----------------
// grid = B * NH * (T/128); 256 threads = 8 waves; wave handles a 16-query tile.
// Q pre-scaled by HD^-0.5. Sink folded into online-softmax init (m0=sink, l0=1).
__global__ __launch_bounds__(256) void flash_attn_kernel(const unsigned short* __restrict__ Qbf, // [B*T][NH*HD]
                                                         const unsigned short* __restrict__ Kbf, // [B*T][NKV*HD]
                                                         const unsigned short* __restrict__ Vt,  // [B][NKV][VHD][T]
                                                         const float* __restrict__ sinkb,        // [NH]
                                                         unsigned short* __restrict__ Obf) {     // [B*T][NH*VHD]
    __shared__ unsigned short ldsP[8][16 * 32];
    const int tid = threadIdx.x, lane = tid & 31, wave = tid >> 5;
    const int half = lane >> 4, l16 = lane & 15;
    const int TILES = TT / 128;
    const int b    = blockIdx.x / (NH * TILES);
    const int rem  = blockIdx.x % (NH * TILES);
    const int h    = rem / TILES;
    const int tile = rem % TILES;
    const int q0   = tile * 128 + wave * 16;
    const int kvh  = h / NREP;

    v16bf qfr[4];
    {
        const unsigned short* qrow = Qbf + (size_t)(b * TT + q0 + l16) * (NH * HD) + h * HD;
#pragma unroll
        for (int d = 0; d < 4; ++d) {
            const v8u* p = (const v8u*)(qrow + d * 32);
            qfr[d] = mk16u(p[2 * half], p[2 * half + 1]);
        }
    }

    float mrow[8], lrow[8];
    v8f acc[8] = {};
    const float sb = sinkb[h];
#pragma unroll
    for (int r = 0; r < 8; ++r) { mrow[r] = sb; lrow[r] = 1.0f; }

    int kmin = q0 - (WIN - 1); if (kmin < 0) kmin = 0; kmin &= ~31;
    const int kmax = q0 + 15;

    for (int kb = kmin; kb <= kmax; kb += 32) {
        v8f S[2] = {};
#pragma unroll
        for (int nh = 0; nh < 2; ++nh) {
            const int key = kb + nh * 16 + l16;
            const unsigned short* krow = Kbf + (size_t)(b * TT + key) * (NKV * HD) + kvh * HD;
#pragma unroll
            for (int d = 0; d < 4; ++d) {
                const v8u* p = (const v8u*)(krow + d * 32 + 16 * half);
                S[nh] = wmma_bf16(qfr[d], mk16u(p[0], p[1]), S[nh]);
            }
        }
#pragma unroll
        for (int r = 0; r < 8; ++r) {
            const int qr = q0 + r + 8 * half;
#pragma unroll
            for (int nh = 0; nh < 2; ++nh) {
                const int key = kb + nh * 16 + l16;
                if (key > qr || key < qr - (WIN - 1)) S[nh][r] = -3.0e38f;
            }
            float mx = fmaxf(S[0][r], S[1][r]);
#pragma unroll
            for (int off = 1; off < 16; off <<= 1) mx = fmaxf(mx, __shfl_xor(mx, off, 32));
            const float nm = fmaxf(mrow[r], mx);
            const float alpha = __expf(mrow[r] - nm);
            const float p0 = __expf(S[0][r] - nm);
            const float p1 = __expf(S[1][r] - nm);
            S[0][r] = p0; S[1][r] = p1;
            float rs = p0 + p1;
#pragma unroll
            for (int off = 1; off < 16; off <<= 1) rs += __shfl_xor(rs, off, 32);
            lrow[r] = lrow[r] * alpha + rs;
            mrow[r] = nm;
#pragma unroll
            for (int c = 0; c < 8; ++c) acc[c][r] *= alpha;
        }
        // relayout P (C-layout) -> A-fragment via wave-private LDS bounce
        unsigned short* Pl = ldsP[wave];
#pragma unroll
        for (int r = 0; r < 8; ++r) {
            const int m = r + 8 * half;
            Pl[m * 32 + l16]      = f2bf(S[0][r]);
            Pl[m * 32 + 16 + l16] = f2bf(S[1][r]);
        }
        asm volatile("s_wait_dscnt 0" ::: "memory");
        v16bf pfr;
        {
            const v8u* rowp = (const v8u*)(Pl + l16 * 32);
            pfr = mk16u(rowp[2 * half], rowp[2 * half + 1]);
        }
#pragma unroll
        for (int c = 0; c < 8; ++c) {
            const unsigned short* vcol =
                Vt + (((size_t)(b * NKV + kvh) * VHD) + c * 16 + l16) * TT + kb + 16 * half;
            const v8u* p = (const v8u*)(vcol);
            acc[c] = wmma_bf16(pfr, mk16u(p[0], p[1]), acc[c]);
        }
    }
#pragma unroll
    for (int c = 0; c < 8; ++c)
#pragma unroll
        for (int r = 0; r < 8; ++r) {
            const float o = acc[c][r] / lrow[r];
            const size_t idx = (size_t)(b * TT + q0 + r + 8 * half) * (NH * VHD)
                             + h * VHD + c * 16 + l16;
            Obf[idx] = f2bf(o);
        }
}

// ---------------- host launch ----------------
static size_t alignUp(size_t v) { return (v + 255) & ~(size_t)255; }

extern "C" void kernel_launch(void* const* d_in, const int* in_sizes, int n_in,
                              void* d_out, int out_size, void* d_ws, size_t ws_size,
                              hipStream_t stream) {
    (void)in_sizes; (void)n_in; (void)out_size; (void)ws_size;
    const float* x        = (const float*)d_in[0];
    const float* wq       = (const float*)d_in[2];
    const float* wk       = (const float*)d_in[3];
    const float* wv       = (const float*)d_in[4];
    const float* wo       = (const float*)d_in[5];
    const float* sinkb    = (const float*)d_in[6];
    const float* gate_w   = (const float*)d_in[7];
    const float* up_w     = (const float*)d_in[8];
    const float* down_w   = (const float*)d_in[9];
    const float* n1_scale = (const float*)d_in[10];
    const float* n2_scale = (const float*)d_in[11];
    float* out = (float*)d_out;

    char* ws = (char*)d_ws;
    size_t off = 0;
    auto take = [&](size_t bytes) { char* p = ws + off; off = alignUp(off + bytes); return p; };
    unsigned short* wq_bf   = (unsigned short*)take((size_t)EMB * NH * HD * 2);
    unsigned short* wk_bf   = (unsigned short*)take((size_t)EMB * NKV * HD * 2);
    unsigned short* wv_bf   = (unsigned short*)take((size_t)EMB * NKV * VHD * 2);
    unsigned short* wo_bf   = (unsigned short*)take((size_t)NH * VHD * EMB * 2);
    unsigned short* gate_bf = (unsigned short*)take((size_t)EMB * MLPD * 2);
    unsigned short* up_bf   = (unsigned short*)take((size_t)EMB * MLPD * 2);
    unsigned short* down_bf = (unsigned short*)take((size_t)MLPD * EMB * 2);
    unsigned short* h1_bf   = (unsigned short*)take((size_t)MBT * EMB * 2);
    float*          qf      = (float*)take((size_t)MBT * NH * HD * 4);
    float*          kf      = (float*)take((size_t)MBT * NKV * HD * 4);
    float*          vf      = (float*)take((size_t)MBT * NKV * VHD * 4);
    unsigned short* q_bf    = (unsigned short*)take((size_t)MBT * NH * HD * 2);
    unsigned short* k_bf    = (unsigned short*)take((size_t)MBT * NKV * HD * 2);
    unsigned short* vt_bf   = (unsigned short*)take((size_t)MBT * NKV * VHD * 2);
    unsigned short* attn_bf = (unsigned short*)take((size_t)MBT * NH * VHD * 2);
    float*          x2      = (float*)take((size_t)MBT * EMB * 4);
    unsigned short* h2_bf   = (unsigned short*)take((size_t)MBT * EMB * 2);
    unsigned short* t_bf    = (unsigned short*)take((size_t)MBT * MLPD * 2);

    auto blocks = [](size_t n) { return (unsigned)((n + 255) / 256); };

    // 1) weights -> bf16
    cast_bf16_kernel<<<blocks((size_t)EMB*NH*HD),   256, 0, stream>>>(wq,     wq_bf,   (size_t)EMB*NH*HD);
    cast_bf16_kernel<<<blocks((size_t)EMB*NKV*HD),  256, 0, stream>>>(wk,     wk_bf,   (size_t)EMB*NKV*HD);
    cast_bf16_kernel<<<blocks((size_t)EMB*NKV*VHD), 256, 0, stream>>>(wv,     wv_bf,   (size_t)EMB*NKV*VHD);
    cast_bf16_kernel<<<blocks((size_t)NH*VHD*EMB),  256, 0, stream>>>(wo,     wo_bf,   (size_t)NH*VHD*EMB);
    cast_bf16_kernel<<<blocks((size_t)EMB*MLPD),    256, 0, stream>>>(gate_w, gate_bf, (size_t)EMB*MLPD);
    cast_bf16_kernel<<<blocks((size_t)EMB*MLPD),    256, 0, stream>>>(up_w,   up_bf,   (size_t)EMB*MLPD);
    cast_bf16_kernel<<<blocks((size_t)MLPD*EMB),    256, 0, stream>>>(down_w, down_bf, (size_t)MLPD*EMB);

    // 2) h1 = rmsnorm(x)
    rmsnorm_bf16_kernel<<<MBT, 256, 0, stream>>>(x, n1_scale, h1_bf);

    // 3) QKV projections
    gemm_kernel<MBT, NH*HD,   EMB, false><<<(MBT/128)*((NH*HD)/128),   256, 0, stream>>>(h1_bf, wq_bf, nullptr, qf);
    gemm_kernel<MBT, NKV*HD,  EMB, false><<<(MBT/128)*((NKV*HD)/128),  256, 0, stream>>>(h1_bf, wk_bf, nullptr, kf);
    gemm_kernel<MBT, NKV*VHD, EMB, false><<<(MBT/128)*((NKV*VHD)/128), 256, 0, stream>>>(h1_bf, wv_bf, nullptr, vf);

    // 4) RoPE + casts; V -> key-contiguous bf16
    rope_cast_kernel<<<blocks((size_t)MBT*NH*HD),  256, 0, stream>>>(qf, q_bf, NH,  SCALING, (size_t)MBT*NH*HD);
    rope_cast_kernel<<<blocks((size_t)MBT*NKV*HD), 256, 0, stream>>>(kf, k_bf, NKV, 1.0f,    (size_t)MBT*NKV*HD);
    v_cast_transpose_kernel<<<blocks((size_t)MBT*NKV*VHD), 256, 0, stream>>>(vf, vt_bf, (size_t)MBT*NKV*VHD);

    // 5) flash attention (windowed causal + sink)
    flash_attn_kernel<<<BB * NH * (TT/128), 256, 0, stream>>>(q_bf, k_bf, vt_bf, sinkb, attn_bf);

    // 6) output projection + residual -> x2
    gemm_kernel<MBT, EMB, NH*VHD, true><<<(MBT/128)*(EMB/128), 256, 0, stream>>>(attn_bf, wo_bf, x, x2);

    // 7) h2 = rmsnorm(x2)
    rmsnorm_bf16_kernel<<<MBT, 256, 0, stream>>>(x2, n2_scale, h2_bf);

    // 8) MLP: fused gate/up/silu, then down + residual -> out
    gateup_silu_kernel<<<(MBT/128)*(MLPD/128), 256, 0, stream>>>(h2_bf, gate_bf, up_bf, t_bf);
    gemm_kernel<MBT, EMB, MLPD, true><<<(MBT/128)*(EMB/128), 256, 0, stream>>>(t_bf, down_bf, x2, out);
}
